// GraphProposalNetwork_87574383165986
// MI455X (gfx1250) — compile-verified, matrix operand order checked by reading, not run on hardware
//
#include <hip/hip_runtime.h>
#include <hip/hip_bf16.h>

// ---- types ----
typedef __attribute__((ext_vector_type(16))) __bf16 v16bf;
typedef __attribute__((ext_vector_type(8)))  float  v8f;
typedef int v4i __attribute__((vector_size(16)));   // matches builtin param type

#define AS1 __attribute__((address_space(1)))
#define AS3 __attribute__((address_space(3)))

#if __has_builtin(__builtin_amdgcn_global_load_async_to_lds_b128) && \
    __has_builtin(__builtin_amdgcn_s_wait_asynccnt)
#define USE_ASYNC_LDS 1
#else
#define USE_ASYNC_LDS 0
#endif

static __device__ __forceinline__ __bf16 f2bf(float f) {
    return (__bf16)f;   // native f32->bf16 RNE
}

static __device__ __forceinline__ v8f wmma_bf16(v16bf a, v16bf b, v8f c) {
    return __builtin_amdgcn_wmma_f32_16x16x32_bf16(
        /*neg_a=*/false, a, /*neg_b=*/false, b,
        /*c_mod=*/(short)0, c, /*reuse_a=*/false, /*reuse_b=*/false);
}

// 16B global -> LDS copy; async (ASYNCcnt) when available, else via VGPRs.
static __device__ __forceinline__ void copy16_g2l(const __bf16* g, __bf16* l) {
#if USE_ASYNC_LDS
    __builtin_amdgcn_global_load_async_to_lds_b128((AS1 v4i*)g, (AS3 v4i*)l, 0, 0);
#else
    *(uint4*)l = *(const uint4*)g;
#endif
}
static __device__ __forceinline__ void g2l_wait() {
#if USE_ASYNC_LDS
    __builtin_amdgcn_s_wait_asynccnt(0);
#endif
}

// K offset held by (lane, j) inside a 32-wide K block, per 16-bit A/B VGPR layout:
// lanes 0-15: j=0..7 -> K 0..7, j=8..15 -> K 16..23 ; lanes 16-31: +8
static __device__ __forceinline__ int koff(int lane, int j) {
    return ((j < 8) ? j : (j + 8)) + ((lane >= 16) ? 8 : 0);
}

// ---------------------------------------------------------------------------
// Prep: pack a row-major f32 weight (K x srcN) into the bf16 per-lane WMMA
// B-operand image: packed[((kb*NT + nt)*32 + lane)*16 + j], col = nt*16+lane%16,
// k = kb*32 + koff(lane,j).  Pads cols >= srcN with zero (Wc3: 10 -> 16).
// ---------------------------------------------------------------------------
__global__ void pack_weight_kernel(const float* __restrict__ W,
                                   __bf16* __restrict__ dst,
                                   int K, int srcN, int dstN) {
    int total = K * dstN;
    int e = blockIdx.x * blockDim.x + threadIdx.x;
    if (e >= total) return;
    int NT   = dstN >> 4;
    int j    = e & 15;
    int lane = (e >> 4) & 31;
    int t    = e >> 9;
    int nt   = t % NT;
    int kb   = t / NT;
    int k    = kb * 32 + koff(lane, j);
    int col  = nt * 16 + (lane & 15);
    float v  = (col < srcN) ? W[(size_t)k * srcN + col] : 0.0f;
    dst[e] = f2bf(v);
}

// ---------------------------------------------------------------------------
// Kernel 1: v = (X@Wp + bp) * gm + ga  ->  V (bf16, [81920 x 256])
// block = 256 threads (8 waves); each wave: 16 rows x 256 cols.
// Per K-step the 16KB B slice is staged once per block into LDS
// (async global->LDS, double buffered) and shared by all 8 waves.
// ---------------------------------------------------------------------------
__global__ void __launch_bounds__(256)
fused_embed_gate_kernel(const float* __restrict__ X,      // [81920 x 1024]
                        const float* __restrict__ G,      // [81920 x 4]
                        const __bf16* __restrict__ Wp_p,  // packed 1024x256
                        const float* __restrict__ bp,
                        const float* __restrict__ Wm1, const float* __restrict__ bm1,
                        const __bf16* __restrict__ Wm2_p, const float* __restrict__ bm2,
                        const float* __restrict__ Wa1, const float* __restrict__ ba1,
                        const __bf16* __restrict__ Wa2_p, const float* __restrict__ ba2,
                        __bf16* __restrict__ V)           // [81920 x 256] bf16
{
    __shared__ __align__(32) __bf16 ldsB[2][8192];   // 2 x 16KB B slices

    const int tid  = threadIdx.x;
    const int wave = tid >> 5;
    const int lane = tid & 31;
    const int m0   = blockIdx.x * 128 + wave * 16;
    const int lrow = lane & 15;
    const bool hi  = lane >= 16;
    const int row  = m0 + lrow;               // A-operand row for this lane

    // stage B slice for K-step ks into buffer buf (256 thr x 64B = 16KB)
    auto issueB = [&](int ks, int buf) {
        const __bf16* src = Wp_p + (size_t)ks * 8192 + tid * 32;
        __bf16* dstp = &ldsB[buf][tid * 32];
        #pragma unroll
        for (int c = 0; c < 4; ++c) copy16_g2l(src + c * 8, dstp + c * 8);
    };

    v8f acc[16];
    #pragma unroll
    for (int nt = 0; nt < 16; ++nt) acc[nt] = (v8f)0.0f;

    issueB(0, 0);

    // ---- main GEMM: K = 1024 in 32 WMMA steps ----
    for (int ks = 0; ks < 32; ++ks) {
        const int cur = ks & 1;
        g2l_wait();
        __syncthreads();                       // slice `cur` visible to all waves
        if (ks < 31) issueB(ks + 1, cur ^ 1);  // overlap next copy with compute

        const int kbase = ks * 32 + (hi ? 8 : 0);
        const float* a0 = X + (size_t)row * 1024 + kbase;
        float4 f0 = *(const float4*)(a0);
        float4 f1 = *(const float4*)(a0 + 4);
        float4 f2 = *(const float4*)(a0 + 16);
        float4 f3 = *(const float4*)(a0 + 20);
        v16bf A;
        A[0]=f2bf(f0.x); A[1]=f2bf(f0.y); A[2]=f2bf(f0.z); A[3]=f2bf(f0.w);
        A[4]=f2bf(f1.x); A[5]=f2bf(f1.y); A[6]=f2bf(f1.z); A[7]=f2bf(f1.w);
        A[8]=f2bf(f2.x); A[9]=f2bf(f2.y); A[10]=f2bf(f2.z); A[11]=f2bf(f2.w);
        A[12]=f2bf(f3.x); A[13]=f2bf(f3.y); A[14]=f2bf(f3.z); A[15]=f2bf(f3.w);

        #pragma unroll
        for (int nt = 0; nt < 16; ++nt) {
            v16bf B = *(const v16bf*)&ldsB[cur][((size_t)nt * 32 + lane) * 16];
            acc[nt] = wmma_bf16(A, B, acc[nt]);
        }
        __syncthreads();                       // all waves done reading `cur`
    }

    // ---- gate hidden layers (4 -> 32), built directly in A-register layout ----
    const float* g = G + (size_t)row * 4;
    const float g0 = g[0], g1 = g[1], g2 = g[2], g3 = g[3];
    v16bf Hm, Ha;
    #pragma unroll
    for (int j = 0; j < 16; ++j) {
        const int jg = koff(lane, j);
        float sm = bm1[jg] + g0 * Wm1[jg] + g1 * Wm1[32 + jg]
                           + g2 * Wm1[64 + jg] + g3 * Wm1[96 + jg];
        sm = (sm >= 0.0f) ? sm : 0.02f * sm;
        Hm[j] = f2bf(sm);
        float sa = ba1[jg] + g0 * Wa1[jg] + g1 * Wa1[32 + jg]
                           + g2 * Wa1[64 + jg] + g3 * Wa1[96 + jg];
        sa = (sa >= 0.0f) ? sa : 0.02f * sa;
        Ha[j] = f2bf(sa);
    }

    // ---- per N-tile: gm/ga WMMA (K=32), combine, store bf16 ----
    const int col   = lane & 15;
    const int rbase = hi ? 8 : 0;           // C-layout row base for this lane
    #pragma unroll
    for (int nt = 0; nt < 16; ++nt) {
        v16bf Bm = *(const v16bf*)(Wm2_p + ((size_t)nt * 32 + lane) * 16);
        v16bf Ba = *(const v16bf*)(Wa2_p + ((size_t)nt * 32 + lane) * 16);
        v8f gm = (v8f)0.0f, ga = (v8f)0.0f;
        gm = wmma_bf16(Hm, Bm, gm);
        ga = wmma_bf16(Ha, Ba, ga);
        const int n = nt * 16 + col;
        const float bpv  = bp[n];
        const float bm2v = bm2[n];
        const float ba2v = ba2[n];
        #pragma unroll
        for (int r = 0; r < 8; ++r) {
            float vv = (acc[nt][r] + bpv) * (gm[r] + bm2v) + (ga[r] + ba2v);
            const int rowg = m0 + rbase + r;
            V[(size_t)rowg * 256 + n] = f2bf(vv);
        }
    }
}

// ---------------------------------------------------------------------------
// Kernel 2: head MLP 512->128->64->10 (+output transpose).
// block = 128 threads (4 waves); each wave: 16 comp rows end-to-end.
// comp row r=b*10+n: [ V[b*10+9] , V[b*10 + (n==0 ? 9 : n-1)] ]
// Stage-1 Wc1 slices staged per block via async global->LDS double buffer.
// ---------------------------------------------------------------------------
__global__ void __launch_bounds__(128)
head_kernel(const __bf16* __restrict__ V,
            const __bf16* __restrict__ Wc1_p, const float* __restrict__ bc1,
            const __bf16* __restrict__ Wc2_p, const float* __restrict__ bc2,
            const __bf16* __restrict__ Wc3_p, const float* __restrict__ bc3,
            float* __restrict__ out)          // [8192 x 10 x 10]
{
    __shared__ __align__(32) __bf16 lds_act[4][16 * 128]; // per-wave act relayout
    __shared__ __align__(32) __bf16 ldsB[2][4096];        // 2 x 8KB Wc1 slices

    const int tid  = threadIdx.x;
    const int wave = tid >> 5;
    const int lane = tid & 31;
    const int m0   = blockIdx.x * 64 + wave * 16;
    const int lrow = lane & 15;
    const bool hi  = lane >= 16;
    const int col  = lane & 15;
    const int rbase = hi ? 8 : 0;
    __bf16* my = lds_act[wave];

    auto issueB = [&](int ks, int buf) {     // 128 thr x 64B = 8KB
        const __bf16* src = Wc1_p + (size_t)ks * 4096 + tid * 32;
        __bf16* dstp = &ldsB[buf][tid * 32];
        #pragma unroll
        for (int c = 0; c < 4; ++c) copy16_g2l(src + c * 8, dstp + c * 8);
    };

    // shifted-concat source rows for this lane's A row
    const int r  = m0 + lrow;
    const int b  = r / 10;
    const int n  = r - b * 10;
    const size_t row_vi = ((size_t)b * 10 + 9) * 256;
    const size_t row_rt = ((size_t)b * 10 + (n == 0 ? 9 : (n - 1))) * 256;

    // ---- stage 1: comp(512) @ Wc1 -> h1 (16 x 128) ----
    v8f h1[8];
    #pragma unroll
    for (int nt = 0; nt < 8; ++nt) h1[nt] = (v8f)0.0f;

    issueB(0, 0);
    for (int ks = 0; ks < 16; ++ks) {
        const int cur = ks & 1;
        g2l_wait();
        __syncthreads();
        if (ks < 15) issueB(ks + 1, cur ^ 1);

        size_t base = (ks < 8) ? (row_vi + ks * 32) : (row_rt + (ks - 8) * 32);
        base += hi ? 8 : 0;
        union { uint4 q[2]; v16bf v; } A;
        A.q[0] = *(const uint4*)(V + base);
        A.q[1] = *(const uint4*)(V + base + 16);
        #pragma unroll
        for (int nt = 0; nt < 8; ++nt) {
            v16bf B = *(const v16bf*)&ldsB[cur][((size_t)nt * 32 + lane) * 16];
            h1[nt] = wmma_bf16(A.v, B, h1[nt]);
        }
        __syncthreads();
    }

    // bias+relu -> LDS (C-layout -> row-major bf16 tile 16x128)
    #pragma unroll
    for (int nt = 0; nt < 8; ++nt) {
        const float bb = bc1[nt * 16 + col];
        #pragma unroll
        for (int rr = 0; rr < 8; ++rr) {
            float x = h1[nt][rr] + bb;
            x = x > 0.0f ? x : 0.0f;
            my[(rbase + rr) * 128 + nt * 16 + col] = f2bf(x);
        }
    }
    __syncthreads();

    // ---- stage 2: h1(128) @ Wc2 -> h2 (16 x 64) ----
    v8f h2[4];
    #pragma unroll
    for (int nt = 0; nt < 4; ++nt) h2[nt] = (v8f)0.0f;
    #pragma unroll
    for (int ks = 0; ks < 4; ++ks) {
        const int kb = ks * 32;
        v16bf A;
        #pragma unroll
        for (int j = 0; j < 16; ++j)
            A[j] = my[lrow * 128 + kb + koff(lane, j)];
        #pragma unroll
        for (int nt = 0; nt < 4; ++nt) {
            v16bf B = *(const v16bf*)(Wc2_p + (((size_t)ks * 4 + nt) * 32 + lane) * 16);
            h2[nt] = wmma_bf16(A, B, h2[nt]);
        }
    }
    __syncthreads();
    #pragma unroll
    for (int nt = 0; nt < 4; ++nt) {
        const float bb = bc2[nt * 16 + col];
        #pragma unroll
        for (int rr = 0; rr < 8; ++rr) {
            float x = h2[nt][rr] + bb;
            x = x > 0.0f ? x : 0.0f;
            my[(rbase + rr) * 64 + nt * 16 + col] = f2bf(x);
        }
    }
    __syncthreads();

    // ---- stage 3: h2(64) @ Wc3(64x16, cols 10..15 zero) -> 16 x 16 ----
    v8f h3 = (v8f)0.0f;
    #pragma unroll
    for (int ks = 0; ks < 2; ++ks) {
        const int kb = ks * 32;
        v16bf A;
        #pragma unroll
        for (int j = 0; j < 16; ++j)
            A[j] = my[lrow * 64 + kb + koff(lane, j)];
        v16bf B = *(const v16bf*)(Wc3_p + ((size_t)ks * 32 + lane) * 16);
        h3 = wmma_bf16(A, B, h3);
    }

    // ---- output: out[b, c, n] = h3 + bc3  (transpose via scatter) ----
    if (col < 10) {
        const float bb = bc3[col];
        #pragma unroll
        for (int rr = 0; rr < 8; ++rr) {
            const int rg = m0 + rbase + rr;
            const int bo = rg / 10;
            const int no = rg - bo * 10;
            out[(size_t)bo * 100 + col * 10 + no] = h3[rr] + bb;
        }
    }
}

// ---------------------------------------------------------------------------
// launcher
// ---------------------------------------------------------------------------
extern "C" void kernel_launch(void* const* d_in, const int* in_sizes, int n_in,
                              void* d_out, int out_size, void* d_ws, size_t ws_size,
                              hipStream_t stream) {
    const float* X   = (const float*)d_in[0];   // object_features  [8192,10,1024]
    const float* G   = (const float*)d_in[1];   // scene_geometry   [8192,10,4]
    const float* Wp  = (const float*)d_in[2];
    const float* bp  = (const float*)d_in[3];
    const float* Wm1 = (const float*)d_in[4];
    const float* bm1 = (const float*)d_in[5];
    const float* Wm2 = (const float*)d_in[6];
    const float* bm2 = (const float*)d_in[7];
    const float* Wa1 = (const float*)d_in[8];
    const float* ba1 = (const float*)d_in[9];
    const float* Wa2 = (const float*)d_in[10];
    const float* ba2 = (const float*)d_in[11];
    const float* Wc1 = (const float*)d_in[12];
    const float* bc1 = (const float*)d_in[13];
    const float* Wc2 = (const float*)d_in[14];
    const float* bc2 = (const float*)d_in[15];
    const float* Wc3 = (const float*)d_in[16];
    const float* bc3 = (const float*)d_in[17];
    float* out = (float*)d_out;

    // workspace layout (bf16 elements)
    __bf16* wsb = (__bf16*)d_ws;
    const size_t OFF_WP  = 0;                       // 1024*256
    const size_t OFF_WM2 = OFF_WP  + 262144;        // 32*256
    const size_t OFF_WA2 = OFF_WM2 + 8192;
    const size_t OFF_WC1 = OFF_WA2 + 8192;          // 512*128
    const size_t OFF_WC2 = OFF_WC1 + 65536;         // 128*64
    const size_t OFF_WC3 = OFF_WC2 + 8192;          // 64*16 (padded)
    const size_t OFF_V   = OFF_WC3 + 1024;          // 81920*256

    __bf16* Wp_p  = wsb + OFF_WP;
    __bf16* Wm2_p = wsb + OFF_WM2;
    __bf16* Wa2_p = wsb + OFF_WA2;
    __bf16* Wc1_p = wsb + OFF_WC1;
    __bf16* Wc2_p = wsb + OFF_WC2;
    __bf16* Wc3_p = wsb + OFF_WC3;
    __bf16* V     = wsb + OFF_V;

    auto pack = [&](const float* W, __bf16* dst, int K, int srcN, int dstN) {
        int total = K * dstN;
        pack_weight_kernel<<<(total + 255) / 256, 256, 0, stream>>>(W, dst, K, srcN, dstN);
    };
    pack(Wp,  Wp_p,  1024, 256, 256);
    pack(Wm2, Wm2_p,   32, 256, 256);
    pack(Wa2, Wa2_p,   32, 256, 256);
    pack(Wc1, Wc1_p,  512, 128, 128);
    pack(Wc2, Wc2_p,  128,  64,  64);
    pack(Wc3, Wc3_p,   64,  10,  16);

    // 81920 rows / 128 rows per block
    fused_embed_gate_kernel<<<640, 256, 0, stream>>>(
        X, G, Wp_p, bp, Wm1, bm1, Wm2_p, bm2, Wa1, ba1, Wa2_p, ba2, V);

    // 81920 rows / 64 rows per block
    head_kernel<<<1280, 128, 0, stream>>>(
        V, Wc1_p, bc1, Wc2_p, bc2, Wc3_p, bc3, out);
}
